// VariableSelectionNetwork_5609227289158
// MI455X (gfx1250) — matrix operand order repaired and checked by reading, not compiled, not run on hardware
//
#include <hip/hip_runtime.h>

// Problem dims
#define BTOT   16384   // B*T
#define HID    256     // hidden
#define VVARS  16      // n_vars
#define DIN    64      // d_inp
#define VD     1024    // VVARS*DIN

typedef __attribute__((ext_vector_type(8)))  __bf16 bf16x8;
typedef __attribute__((ext_vector_type(16))) __bf16 bf16x16;
typedef __attribute__((ext_vector_type(8)))  float  f32x8;

// ---- helpers -------------------------------------------------------------

// Load A-operand 16 bf16 values for one 16x32 K-block.
// Caller passes pointer already offset to (row, kb*32 + half*8).
static __device__ __forceinline__ bf16x16 load_a16(const __bf16* p) {
  bf16x8 lo = *(const bf16x8*)p;         // K = base .. base+7
  bf16x8 hi = *(const bf16x8*)(p + 16);  // K = base+16 .. base+23
  bf16x16 a;
#pragma unroll
  for (int i = 0; i < 8; ++i) { a[i] = lo[i]; a[8 + i] = hi[i]; }
  return a;
}

static __device__ __forceinline__ float elu1(float x) {
  return x > 0.f ? x : (__expf(x) - 1.f);
}
static __device__ __forceinline__ float sigm(float x) {
  return 1.f / (1.f + __expf(-x));
}

// ---- fp32 -> bf16 convert ------------------------------------------------
__global__ void k_cvt_bf16(const float* __restrict__ src, __bf16* __restrict__ dst, int n) {
  int i = blockIdx.x * blockDim.x + threadIdx.x;
  if (i < n) dst[i] = (__bf16)src[i];
}

// ---- pack weight matrices [cnt][K][N] (row-major) into wave32 WMMA B layout:
// packed idx = ((nt*KB + kb)*32 + lane)*16 + i  <->  K = kb*32 + (lane>>4)*8 + (i&7) + (i>>3)*16,
//                                                    N = nt*16 + (lane&15)
__global__ void k_pack_b(const float* __restrict__ src, __bf16* __restrict__ dst,
                         int K, int N, int total) {
  int tid = blockIdx.x * blockDim.x + threadIdx.x;
  if (tid >= total) return;
  int kn = K * N;
  int m = tid / kn;
  int r = tid - m * kn;
  int i    = r & 15;
  int lane = (r >> 4) & 31;
  int t    = r >> 9;
  int KB = K >> 5;
  int kb = t % KB;
  int nt = t / KB;
  int k = kb * 32 + ((lane >> 4) << 3) + (i & 7) + ((i >> 3) << 4);
  int n = (nt << 4) + (lane & 15);
  dst[tid] = (__bf16)src[(size_t)m * kn + (size_t)k * N + n];
}

// ---- weight-GRN fc1: h = elu(flat @ W1 + b1)   [BT,1024]x[1024,256] ------
__global__ void __launch_bounds__(256)
k_fc1(const __bf16* __restrict__ xb, const __bf16* __restrict__ pB,
      const float* __restrict__ bias, __bf16* __restrict__ hout) {
  const int l    = threadIdx.x;
  const int nt   = blockIdx.y * 8 + threadIdx.y;   // 0..15 N-tile
  const int m0   = blockIdx.x * 16;
  const int row  = l & 15, half = l >> 4;
  const int c    = nt * 16 + (l & 15);
  const int KB   = VD / 32;
  const __bf16* arow = xb + (size_t)(m0 + row) * VD + half * 8;
  const __bf16* bp   = pB + ((size_t)nt * KB) * 512 + (size_t)l * 16;
  f32x8 acc = {};
#pragma unroll 4
  for (int kb = 0; kb < KB; ++kb) {
    bf16x16 A = load_a16(arow + kb * 32);
    bf16x16 B = *(const bf16x16*)(bp + (size_t)kb * 512);
    acc = __builtin_amdgcn_wmma_f32_16x16x32_bf16(false, A, false, B, (short)0, acc, false, false);
  }
  float b = bias[c];
#pragma unroll
  for (int j = 0; j < 8; ++j)
    hout[(size_t)(m0 + j + half * 8) * HID + c] = (__bf16)elu1(acc[j] + b);
}

// ---- selection weights: y=h@W2+b2; gate=sig(y@Wg+bg); res=flat@Ws+bs;
//      wsel = softmax(res + gate*y)  over the V=16 axis ---------------------
__global__ void __launch_bounds__(256)
k_sel(const __bf16* __restrict__ xb, const __bf16* __restrict__ hb,
      const __bf16* __restrict__ pFc2, const __bf16* __restrict__ pSkip,
      const float* __restrict__ fc2_b, const float* __restrict__ gate_w,
      const float* __restrict__ gate_b, const float* __restrict__ skip_b,
      float* __restrict__ wsel) {
  __shared__ float yl[8][16][16];
  const int l = threadIdx.x, wy = threadIdx.y;
  const int m0  = (blockIdx.x * 8 + wy) * 16;
  const int row = l & 15, half = l >> 4;
  const int c   = l & 15;

  // y = h @ fc2 + b  (K=256, N=16)
  {
    const __bf16* arow = hb + (size_t)(m0 + row) * HID + half * 8;
    const __bf16* bp   = pFc2 + (size_t)l * 16;
    f32x8 y = {};
#pragma unroll
    for (int kb = 0; kb < 8; ++kb) {
      bf16x16 A = load_a16(arow + kb * 32);
      bf16x16 B = *(const bf16x16*)(bp + (size_t)kb * 512);
      y = __builtin_amdgcn_wmma_f32_16x16x32_bf16(false, A, false, B, (short)0, y, false, false);
    }
    float b = fc2_b[c];
#pragma unroll
    for (int j = 0; j < 8; ++j) yl[wy][j + half * 8][c] = y[j] + b;
  }

  // res = flat @ skip (K=1024, N=16)
  f32x8 res = {};
  {
    const __bf16* arow = xb + (size_t)(m0 + row) * VD + half * 8;
    const __bf16* bp   = pSkip + (size_t)l * 16;
#pragma unroll 4
    for (int kb = 0; kb < 32; ++kb) {
      bf16x16 A = load_a16(arow + kb * 32);
      bf16x16 B = *(const bf16x16*)(bp + (size_t)kb * 512);
      res = __builtin_amdgcn_wmma_f32_16x16x32_bf16(false, A, false, B, (short)0, res, false, false);
    }
  }
  __syncthreads();

  const float bs = skip_b[c];
  const float bg = gate_b[c];
  float s[8];
#pragma unroll
  for (int j = 0; j < 8; ++j) {
    int r = j + half * 8;
    float g = bg;
#pragma unroll
    for (int k = 0; k < 16; ++k) g += yl[wy][r][k] * gate_w[k * 16 + c];
    g = sigm(g);
    s[j] = res[j] + bs + g * yl[wy][r][c];
  }
  // softmax across the 16 columns (lanes of this half-wave)
#pragma unroll
  for (int j = 0; j < 8; ++j) {
    float mx = s[j];
#pragma unroll
    for (int off = 1; off < 16; off <<= 1) mx = fmaxf(mx, __shfl_xor(mx, off, 32));
    float e = __expf(s[j] - mx);
    float sm = e;
#pragma unroll
    for (int off = 1; off < 16; off <<= 1) sm += __shfl_xor(sm, off, 32);
    wsel[(size_t)(m0 + j + half * 8) * VVARS + c] = e / sm;
  }
}

// ---- fused per-variable GRNs + weighted sum -------------------------------
// block = 16 waves; wave nt owns output columns [16nt,16nt+16); block owns 16 rows.
// Loops v=0..15, staging hv/yv bf16 tiles in LDS in A-packed layout.
__global__ void __launch_bounds__(512)
k_pervar(const __bf16* __restrict__ xb,
         const __bf16* __restrict__ pVfc1, const __bf16* __restrict__ pVfc2,
         const __bf16* __restrict__ pVgate, const __bf16* __restrict__ pVskip,
         const float* __restrict__ vfc1_b, const float* __restrict__ vfc2_b,
         const float* __restrict__ vgate_b, const float* __restrict__ vskip_b,
         const float* __restrict__ wsel, float* __restrict__ zout) {
  __shared__ __align__(32) __bf16 hvA[8 * 512];  // 8 K-blocks x 32 lanes x 16 bf16
  __shared__ __align__(32) __bf16 yvA[8 * 512];
  const int l   = threadIdx.x;
  const int nt  = threadIdx.y;                 // 0..15
  const int m0  = blockIdx.x * 16;
  const int row = l & 15, half = l >> 4;
  const int c   = nt * 16 + (l & 15);          // output column == K-index of next GEMM
  // scatter coordinates for writing value with K=c into A-packed LDS
  const int kb2   = c >> 5;
  const int kk    = c & 31;
  const int half2 = (kk >> 3) & 1;
  const int slot  = (kk & 7) + (kk >= 16 ? 8 : 0);

  f32x8 z = {};
  for (int v = 0; v < VVARS; ++v) {
    const __bf16* xrow = xb + (size_t)(m0 + row) * VD + v * DIN + half * 8;
    bf16x16 xA0 = load_a16(xrow);
    bf16x16 xA1 = load_a16(xrow + 32);

    // hv = elu(x_v @ v_fc1[v] + b)   K=64
    {
      const __bf16* bp = pVfc1 + (((size_t)v * 16 + nt) * 2) * 512 + (size_t)l * 16;
      f32x8 hv = {};
      hv = __builtin_amdgcn_wmma_f32_16x16x32_bf16(false, xA0, false, *(const bf16x16*)bp,         (short)0, hv, false, false);
      hv = __builtin_amdgcn_wmma_f32_16x16x32_bf16(false, xA1, false, *(const bf16x16*)(bp + 512), (short)0, hv, false, false);
      float b = vfc1_b[v * HID + c];
#pragma unroll
      for (int j = 0; j < 8; ++j) {
        int lane2 = half2 * 16 + (j + half * 8);
        hvA[kb2 * 512 + lane2 * 16 + slot] = (__bf16)elu1(hv[j] + b);
      }
    }
    __syncthreads();

    // yv = hv @ v_fc2[v] + b   K=256
    f32x8 yv = {};
    {
      const __bf16* bp = pVfc2 + (((size_t)v * 16 + nt) * 8) * 512 + (size_t)l * 16;
#pragma unroll
      for (int kb = 0; kb < 8; ++kb) {
        bf16x16 A = *(const bf16x16*)&hvA[kb * 512 + l * 16];
        yv = __builtin_amdgcn_wmma_f32_16x16x32_bf16(false, A, false, *(const bf16x16*)(bp + (size_t)kb * 512), (short)0, yv, false, false);
      }
      float b = vfc2_b[v * HID + c];
#pragma unroll
      for (int j = 0; j < 8; ++j) {
        float y = yv[j] + b;
        yv[j] = y;
        int lane2 = half2 * 16 + (j + half * 8);
        yvA[kb2 * 512 + lane2 * 16 + slot] = (__bf16)y;
      }
    }
    __syncthreads();

    // gv = sigmoid(yv @ v_gate[v] + b)   K=256
    f32x8 gv = {};
    {
      const __bf16* bp = pVgate + (((size_t)v * 16 + nt) * 8) * 512 + (size_t)l * 16;
#pragma unroll
      for (int kb = 0; kb < 8; ++kb) {
        bf16x16 A = *(const bf16x16*)&yvA[kb * 512 + l * 16];
        gv = __builtin_amdgcn_wmma_f32_16x16x32_bf16(false, A, false, *(const bf16x16*)(bp + (size_t)kb * 512), (short)0, gv, false, false);
      }
    }
    const float bg = vgate_b[v * HID + c];

    // rv = x_v @ v_skip[v] + b   K=64
    f32x8 rv = {};
    {
      const __bf16* bp = pVskip + (((size_t)v * 16 + nt) * 2) * 512 + (size_t)l * 16;
      rv = __builtin_amdgcn_wmma_f32_16x16x32_bf16(false, xA0, false, *(const bf16x16*)bp,         (short)0, rv, false, false);
      rv = __builtin_amdgcn_wmma_f32_16x16x32_bf16(false, xA1, false, *(const bf16x16*)(bp + 512), (short)0, rv, false, false);
    }
    const float br = vskip_b[v * HID + c];

#pragma unroll
    for (int j = 0; j < 8; ++j) {
      float wv = wsel[(size_t)(m0 + j + half * 8) * VVARS + v];
      float g  = sigm(gv[j] + bg);
      z[j] += wv * (rv[j] + br + g * yv[j]);
    }
  }
#pragma unroll
  for (int j = 0; j < 8; ++j)
    zout[(size_t)(m0 + j + half * 8) * HID + c] = z[j];
}

// ---- launcher -------------------------------------------------------------
extern "C" void kernel_launch(void* const* d_in, const int* in_sizes, int n_in,
                              void* d_out, int out_size, void* d_ws, size_t ws_size,
                              hipStream_t stream) {
  (void)in_sizes; (void)n_in; (void)out_size; (void)ws_size;
  const float* x      = (const float*)d_in[0];
  const float* wfc1w  = (const float*)d_in[1];
  const float* wfc1b  = (const float*)d_in[2];
  const float* wfc2w  = (const float*)d_in[3];
  const float* wfc2b  = (const float*)d_in[4];
  const float* wgatew = (const float*)d_in[5];
  const float* wgateb = (const float*)d_in[6];
  const float* wskipw = (const float*)d_in[7];
  const float* wskipb = (const float*)d_in[8];
  const float* vfc1w  = (const float*)d_in[9];
  const float* vfc1b  = (const float*)d_in[10];
  const float* vfc2w  = (const float*)d_in[11];
  const float* vfc2b  = (const float*)d_in[12];
  const float* vgatew = (const float*)d_in[13];
  const float* vgateb = (const float*)d_in[14];
  const float* vskipw = (const float*)d_in[15];
  const float* vskipb = (const float*)d_in[16];
  float* z = (float*)d_out;

  char* w = (char*)d_ws;
  size_t off = 0;
  auto alloc = [&](size_t bytes) {
    void* p = w + off;
    off += (bytes + 255) & ~(size_t)255;
    return p;
  };
  __bf16* xb     = (__bf16*)alloc((size_t)BTOT * VD * 2);
  __bf16* hbuf   = (__bf16*)alloc((size_t)BTOT * HID * 2);
  float*  wsel   = (float*) alloc((size_t)BTOT * VVARS * 4);
  __bf16* pFc1   = (__bf16*)alloc((size_t)VD * HID * 2);
  __bf16* pFc2   = (__bf16*)alloc((size_t)HID * 16 * 2);
  __bf16* pSkip  = (__bf16*)alloc((size_t)VD * 16 * 2);
  __bf16* pVfc1  = (__bf16*)alloc((size_t)VVARS * DIN * HID * 2);
  __bf16* pVfc2  = (__bf16*)alloc((size_t)VVARS * HID * HID * 2);
  __bf16* pVgate = (__bf16*)alloc((size_t)VVARS * HID * HID * 2);
  __bf16* pVskip = (__bf16*)alloc((size_t)VVARS * DIN * HID * 2);

  {
    int n = BTOT * VD;
    k_cvt_bf16<<<(n + 255) / 256, 256, 0, stream>>>(x, xb, n);
  }
  auto pack = [&](const float* src, __bf16* dst, int cnt, int K, int N) {
    int total = cnt * K * N;
    k_pack_b<<<(total + 255) / 256, 256, 0, stream>>>(src, dst, K, N, total);
  };
  pack(wfc1w, pFc1, 1, VD, HID);
  pack(wfc2w, pFc2, 1, HID, 16);
  pack(wskipw, pSkip, 1, VD, 16);
  pack(vfc1w, pVfc1, VVARS, DIN, HID);
  pack(vfc2w, pVfc2, VVARS, HID, HID);
  pack(vgatew, pVgate, VVARS, HID, HID);
  pack(vskipw, pVskip, VVARS, DIN, HID);

  k_fc1<<<dim3(BTOT / 16, 2), dim3(32, 8), 0, stream>>>(xb, pFc1, wfc1b, hbuf);
  k_sel<<<dim3(BTOT / 128), dim3(32, 8), 0, stream>>>(xb, hbuf, pFc2, pSkip,
                                                      wfc2b, wgatew, wgateb, wskipb, wsel);
  k_pervar<<<dim3(BTOT / 16), dim3(32, 16), 0, stream>>>(xb, pVfc1, pVfc2, pVgate, pVskip,
                                                         vfc1b, vfc2b, vgateb, vskipb, wsel, z);
}
